// UV_Aggregator_7421703487508
// MI455X (gfx1250) — matrix operand order repaired and checked by reading, not compiled, not run on hardware
//
#include <hip/hip_runtime.h>

// ---------------------------------------------------------------------------
// UV_Aggregator for MI455X (gfx1250, wave32, WMMA)
// Dominant cost: TextCNN conv (~224 GFLOP) -> implicit GEMM via
// v_wmma_f32_16x16x32_f16 with f32 accumulation.
// Round 2: 8 waves/block, software-pipelined (double-buffered) A/B fragment
// loads so WMMAs overlap with global/LDS latency instead of
// load -> s_wait_loadcnt 0 -> wmma serialization.
// ---------------------------------------------------------------------------

typedef _Float16 v16h __attribute__((ext_vector_type(16)));
typedef _Float16 v8h  __attribute__((ext_vector_type(8)));
typedef float    v8f  __attribute__((ext_vector_type(8)));

#define N_    256
#define H_    20
#define S_    64
#define D_    64
#define DW_   300
#define FN_   100
#define FPAD  112            // filters padded to 7 x 16
#define DPAD  320            // embedding dim padded to 10 x 32
#define NSEQ  (N_ * H_)      // 5120
#define FEAT  (2 * D_ + 300) // 428
#define NDC   (DPAD / 32)    // 10 K-chunks of 32

// d_ws layout (element = _Float16 for weights, then float for e_w scratch)
#define WP3_OFF   0
#define WP4_OFF   (3 * FPAD * DPAD)                // 107520 halfs
#define WP5_OFF   (WP4_OFF + 4 * FPAD * DPAD)      // 250880 halfs
#define WP_TOTAL  (WP5_OFF + 5 * FPAD * DPAD)      // 430080 halfs
#define EW_BYTE_OFF (WP_TOTAL * 2)                 // 860160 B (256-aligned)

// ---------------------------------------------------------------------------
// K0: repack conv weights -> f16, layout Wp[f][w][d] with f->112, d->320 pad
// ---------------------------------------------------------------------------
__global__ void prepack_weights(const float* __restrict__ w3,
                                const float* __restrict__ w4,
                                const float* __restrict__ w5,
                                _Float16* __restrict__ wp) {
    int idx = blockIdx.x * blockDim.x + threadIdx.x;
    const int s3 = 3 * FPAD * DPAD, s4 = 4 * FPAD * DPAD, s5 = 5 * FPAD * DPAD;
    if (idx >= s3 + s4 + s5) return;
    const float* src; int k, local, off;
    if (idx < s3)            { src = w3; k = 3; local = idx;           off = WP3_OFF; }
    else if (idx < s3 + s4)  { src = w4; k = 4; local = idx - s3;      off = WP4_OFF; }
    else                     { src = w5; k = 5; local = idx - s3 - s4; off = WP5_OFF; }
    int f = local / (k * DPAD);
    int r = local - f * (k * DPAD);
    int w = r / DPAD;
    int d = r - w * DPAD;
    float v = (f < FN_ && d < DW_) ? src[(f * k + w) * DW_ + d] : 0.0f;
    wp[off + local] = (_Float16)v;
}

union Frag { v16h v; v8h h[2]; };

// ---------------------------------------------------------------------------
// Pipelined WMMA chain for one filter height K.
// Wave owns M-tile m and NT filter N-tiles starting at nbase.
// One-step-ahead prefetch: loads for step+1 are issued before the WMMAs of
// step, so each wait only blocks on loads launched a full step earlier.
// ---------------------------------------------------------------------------
template <int K, int NT>
__device__ __forceinline__
void conv_mm(const _Float16* __restrict__ sseq,  // LDS [68][DPAD] f16
             const _Float16* __restrict__ wp,    // packed weights for K
             float*          __restrict__ spmax, // LDS [16][FPAD]
             int m, int nbase, int wv, int lane) {
    v8f acc[NT] = {};

    const int rbase = 16 * m + (lane & 15);   // A row (matrix M = lane&15)
    const int aoff  = (lane >> 4) << 3;       // A: +8 K for upper lanes
    const int boff  = (lane >> 4) << 3;       // B: +8 K for upper lanes
    const int fl    = lane & 15;              // B col within N-tile

    const _Float16* wrow[NT];
    #pragma unroll
    for (int n = 0; n < NT; ++n)
        wrow[n] = wp + (size_t)(16 * (nbase + n) + fl) * (K * DPAD) + boff;

    constexpr int NSTEP = K * NDC;

    Frag a_cur = {}, a_nxt = {};
    Frag b_cur[NT] = {}, b_nxt[NT] = {};

    // prologue: step 0
    {
        const _Float16* arow = sseq + rbase * DPAD + aoff;
        a_cur.h[0] = *(const v8h*)(arow);
        a_cur.h[1] = *(const v8h*)(arow + 16);
        #pragma unroll
        for (int n = 0; n < NT; ++n) {
            const _Float16* p = wrow[n];
            b_cur[n].h[0] = *(const v8h*)(p);
            b_cur[n].h[1] = *(const v8h*)(p + 16);
        }
    }

    #pragma unroll
    for (int step = 0; step < NSTEP; ++step) {
        if (step + 1 < NSTEP) {
            const int w  = (step + 1) / NDC;
            const int dc = (step + 1) - w * NDC;
            const _Float16* arow = sseq + (rbase + w) * DPAD + dc * 32 + aoff;
            a_nxt.h[0] = *(const v8h*)(arow);
            a_nxt.h[1] = *(const v8h*)(arow + 16);
            #pragma unroll
            for (int n = 0; n < NT; ++n) {
                const _Float16* p = wrow[n] + w * DPAD + dc * 32;
                b_nxt[n].h[0] = *(const v8h*)(p);
                b_nxt[n].h[1] = *(const v8h*)(p + 16);
            }
        }
        #pragma unroll
        for (int n = 0; n < NT; ++n)
            acc[n] = __builtin_amdgcn_wmma_f32_16x16x32_f16(
                false, a_cur.v, false, b_cur[n].v, (short)0, acc[n], false, false);
        a_cur = a_nxt;
        #pragma unroll
        for (int n = 0; n < NT; ++n) b_cur[n] = b_nxt[n];
    }

    // Max over valid positions p <= S-K (relu/bias commute with max).
    // C layout: lane L col N=L&15; VGPR j row M=j+8*(L>=16).
    const int half  = lane >> 4;
    const int pbase = 16 * m + 8 * half;
    #pragma unroll
    for (int n = 0; n < NT; ++n) {
        float mx = -3.0e38f;
        #pragma unroll
        for (int j = 0; j < 8; ++j) {
            float v = acc[n][j];
            mx = (pbase + j <= S_ - K) ? fmaxf(mx, v) : mx;
        }
        spmax[(wv * 2 + half) * FPAD + 16 * (nbase + n) + fl] = mx;
    }
}

template <int K>
__device__ __forceinline__
void conv_one(const _Float16* __restrict__ sseq,
              const _Float16* __restrict__ wp,
              const float*    __restrict__ bias,
              float*          __restrict__ ewout,
              float*          __restrict__ spmax,
              int wv, int lane, int tid) {
    const int m  = wv & 3;
    const int nh = wv >> 2;
    if (nh == 0) conv_mm<K, 4>(sseq, wp, spmax, m, 0, wv, lane); // f 0..63
    else         conv_mm<K, 3>(sseq, wp, spmax, m, 4, wv, lane); // f 64..111
    __syncthreads();
    if (tid < FN_) {
        const int rb = (tid < 64) ? 0 : 8;  // waves 0-3 cover f<64, 4-7 rest
        float mmax = -3.0e38f;
        #pragma unroll
        for (int r = 0; r < 8; ++r) mmax = fmaxf(mmax, spmax[(rb + r) * FPAD + tid]);
        ewout[tid] = fmaxf(mmax + bias[tid], 0.0f);
    }
    __syncthreads();
}

// ---------------------------------------------------------------------------
// K1: per-sequence TextCNN. 256 threads = 8 wave32; one block per (n,h).
// ---------------------------------------------------------------------------
__global__ __launch_bounds__(256)
void conv_textcnn(const int*   __restrict__ history_w,
                  const float* __restrict__ word_emb,
                  const float* __restrict__ b3,
                  const float* __restrict__ b4,
                  const float* __restrict__ b5,
                  const _Float16* __restrict__ wpack,
                  float* __restrict__ ew) {
    __shared__ __align__(16) _Float16 sseq[68 * DPAD]; // 43.5 KB
    __shared__ int   swidx[S_];
    __shared__ float spmax[16 * FPAD];                 // 7 KB

    const int s   = blockIdx.x;
    const int tid = threadIdx.x;

    if (tid < S_) swidx[tid] = history_w[s * S_ + tid];
    __syncthreads();

    // Stage gathered sequence into LDS as f16, zero-padded to 68 x 320.
    for (int i = tid; i < 68 * DPAD; i += 256) {
        int row = i / DPAD, d = i - row * DPAD;
        float v = 0.0f;
        if (row < S_ && d < DW_) v = word_emb[(size_t)swidx[row] * DW_ + d];
        sseq[i] = (_Float16)v;
    }
    __syncthreads();

    const int wv = tid >> 5, lane = tid & 31;
    float* ewrow = ew + (size_t)s * 300;
    conv_one<3>(sseq, wpack + WP3_OFF, b3, ewrow + 0,   spmax, wv, lane, tid);
    conv_one<4>(sseq, wpack + WP4_OFF, b4, ewrow + 100, spmax, wv, lane, tid);
    conv_one<5>(sseq, wpack + WP5_OFF, b5, ewrow + 200, spmax, wv, lane, tid);
}

// ---------------------------------------------------------------------------
// K2: MLP chain + attention softmax + aggregation. One block per node n.
// Tiny FLOPs -> plain f32 VALU.
// ---------------------------------------------------------------------------
__global__ __launch_bounds__(256)
void mlp_att(const int*   __restrict__ nodes,
             const int*   __restrict__ history_uv,
             const int*   __restrict__ history_r,
             const float* __restrict__ v2e,
             const float* __restrict__ u2e,
             const float* __restrict__ r2e,
             const float* __restrict__ ew,
             const float* __restrict__ w1,  const float* __restrict__ b1,
             const float* __restrict__ w2,  const float* __restrict__ b2,
             const float* __restrict__ a1W, const float* __restrict__ a1b,
             const float* __restrict__ a2W, const float* __restrict__ a2b,
             const float* __restrict__ a3W, const float* __restrict__ a3b,
             float* __restrict__ out) {
    __shared__ float feat[H_ * FEAT];   // 34.2 KB
    __shared__ float hbuf[H_ * D_];
    __shared__ float obuf[H_ * D_];
    __shared__ float abuf[H_ * D_];
    __shared__ float uvec[D_];
    __shared__ int   suv[H_], sr[H_];
    __shared__ float slog[H_], satt[H_];

    const int n = blockIdx.x, tid = threadIdx.x;

    if (tid < H_) { suv[tid] = history_uv[n * H_ + tid]; sr[tid] = history_r[n * H_ + tid]; }
    if (tid < D_) uvec[tid] = u2e[(size_t)nodes[n] * D_ + tid];
    __syncthreads();

    // feat = [e_uv | e_r | e_w]  (20 x 428)
    for (int i = tid; i < H_ * FEAT; i += 256) {
        int h = i / FEAT, c = i - h * FEAT;
        float v;
        if (c < D_)            v = v2e[(size_t)suv[h] * D_ + c];
        else if (c < 2 * D_)   v = r2e[(size_t)sr[h] * D_ + (c - D_)];
        else                   v = ew[((size_t)n * H_ + h) * 300 + (c - 2 * D_)];
        feat[i] = v;
    }
    __syncthreads();

    // h = relu(feat @ W1^T + b1)
    for (int i = tid; i < H_ * D_; i += 256) {
        int h = i >> 6, d = i & 63;
        const float* wr = w1 + d * FEAT;
        const float* xr = feat + h * FEAT;
        float acc = b1[d];
        for (int f = 0; f < FEAT; ++f) acc += xr[f] * wr[f];
        hbuf[i] = fmaxf(acc, 0.0f);
    }
    __syncthreads();

    // o = relu(h @ W2^T + b2)
    for (int i = tid; i < H_ * D_; i += 256) {
        int h = i >> 6, d = i & 63;
        const float* wr = w2 + d * D_;
        const float* xr = hbuf + h * D_;
        float acc = b2[d];
        for (int f = 0; f < D_; ++f) acc += xr[f] * wr[f];
        obuf[i] = fmaxf(acc, 0.0f);
    }
    __syncthreads();

    // a = relu(concat(o,u) @ A1^T + a1b)
    for (int i = tid; i < H_ * D_; i += 256) {
        int h = i >> 6, d = i & 63;
        const float* wr = a1W + d * (2 * D_);
        const float* xr = obuf + h * D_;
        float acc = a1b[d];
        for (int f = 0; f < D_; ++f) acc += xr[f] * wr[f];
        for (int f = 0; f < D_; ++f) acc += uvec[f] * wr[D_ + f];
        abuf[i] = fmaxf(acc, 0.0f);
    }
    __syncthreads();

    // a = relu(a @ A2^T + a2b)   (reuse hbuf)
    for (int i = tid; i < H_ * D_; i += 256) {
        int h = i >> 6, d = i & 63;
        const float* wr = a2W + d * D_;
        const float* xr = abuf + h * D_;
        float acc = a2b[d];
        for (int f = 0; f < D_; ++f) acc += xr[f] * wr[f];
        hbuf[i] = fmaxf(acc, 0.0f);
    }
    __syncthreads();

    if (tid < H_) {
        float acc = a3b[0];
        const float* xr = hbuf + tid * D_;
        for (int d = 0; d < D_; ++d) acc += xr[d] * a3W[d];
        slog[tid] = acc;
    }
    __syncthreads();

    if (tid == 0) {
        float mx = -3.0e38f;
        for (int h = 0; h < H_; ++h) mx = fmaxf(mx, slog[h]);
        float ss = 0.0f;
        for (int h = 0; h < H_; ++h) { float e = expf(slog[h] - mx); satt[h] = e; ss += e; }
        float inv = 1.0f / ss;
        for (int h = 0; h < H_; ++h) satt[h] *= inv;
    }
    __syncthreads();

    if (tid < D_) {
        float acc = 0.0f;
        for (int h = 0; h < H_; ++h) acc += obuf[h * D_ + tid] * satt[h];
        out[(size_t)n * D_ + tid] = acc;
    }
}

// ---------------------------------------------------------------------------
extern "C" void kernel_launch(void* const* d_in, const int* in_sizes, int n_in,
                              void* d_out, int out_size, void* d_ws, size_t ws_size,
                              hipStream_t stream) {
    const int*   nodes      = (const int*)d_in[0];
    const int*   history_uv = (const int*)d_in[1];
    const int*   history_r  = (const int*)d_in[2];
    const int*   history_w  = (const int*)d_in[3];
    const float* v2e        = (const float*)d_in[4];
    const float* u2e        = (const float*)d_in[5];
    const float* r2e        = (const float*)d_in[6];
    const float* word_emb   = (const float*)d_in[7];
    const float* w3  = (const float*)d_in[8];
    const float* b3  = (const float*)d_in[9];
    const float* w4  = (const float*)d_in[10];
    const float* b4  = (const float*)d_in[11];
    const float* w5  = (const float*)d_in[12];
    const float* b5  = (const float*)d_in[13];
    const float* w1  = (const float*)d_in[14];
    const float* b1  = (const float*)d_in[15];
    const float* w2  = (const float*)d_in[16];
    const float* b2  = (const float*)d_in[17];
    const float* a1W = (const float*)d_in[18];
    const float* a1b = (const float*)d_in[19];
    const float* a2W = (const float*)d_in[20];
    const float* a2b = (const float*)d_in[21];
    const float* a3W = (const float*)d_in[22];
    const float* a3b = (const float*)d_in[23];

    _Float16* wpack = (_Float16*)d_ws;
    float*    ew    = (float*)((char*)d_ws + EW_BYTE_OFF);
    float*    out   = (float*)d_out;

    prepack_weights<<<(WP_TOTAL + 255) / 256, 256, 0, stream>>>(w3, w4, w5, wpack);
    conv_textcnn<<<NSEQ, 256, 0, stream>>>(history_w, word_emb, b3, b4, b5, wpack, ew);
    mlp_att<<<N_, 256, 0, stream>>>(nodes, history_uv, history_r, v2e, u2e, r2e, ew,
                                    w1, b1, w2, b2, a1W, a1b, a2W, a2b, a3W, a3b, out);
}
